// PlainDeltaRule_35588099014912
// MI455X (gfx1250) — matrix operand order repaired and checked by reading, not compiled, not run on hardware
//
#include <hip/hip_runtime.h>
#include <hip/hip_bf16.h>
#include <stdint.h>

// ---------------------------------------------------------------------------
// PlainDeltaRule on MI455X (gfx1250, wave32, WMMA).
//   Kernel P: one-time build of B-swizzled f16 weight images in workspace.
//   Kernel A: token MLP + LayerNorm -> hn (f16) in workspace.
//   Kernel B: chunked delta-rule scan (chunk 32 == WMMA K) + output head.
//             hn chunks are double-buffered via GLOBAL_LOAD_ASYNC_TO_LDS_B128
//             (ASYNCcnt): chunk c+1 streams into LDS while chunk c computes.
// All GEMMs via v_wmma_f32_16x16x32_f16 (f32 accumulate).
// ---------------------------------------------------------------------------

constexpr int kH  = 64;    // hidden
constexpr int kFF = 128;   // ffn
constexpr int kV  = 64;    // vocab
constexpr int kB  = 256;   // batch
constexpr int kL  = 4096;  // sequence length
constexpr int kChunk = 32; // delta-rule chunk size (== f16 WMMA K)

typedef __attribute__((ext_vector_type(16))) _Float16 v16h;
typedef __attribute__((ext_vector_type(8)))  _Float16 v8h;
typedef __attribute__((ext_vector_type(8)))  float    v8f;

union V16U { v16h v; v8h h[2]; };

static __device__ __forceinline__ int lane_id() { return (int)(threadIdx.x & 31u); }

static __device__ __forceinline__ v8f wmma32(v16h a, v16h b, v8f c) {
  // (neg_a, A, neg_b, B, c_mod, C, reuse_a, reuse_b)
  return __builtin_amdgcn_wmma_f32_16x16x32_f16(false, a, false, b, (short)0, c,
                                                false, false);
}

// A operand: 16x32 (MxK) f16 tile from row-major LDS (row stride ld halfs).
// Layout: lanes 0-15 -> M=0..15, K kb..kb+15 (kb = 16*(l>=16)).
static __device__ __forceinline__ v16h lds_load_A16(const _Float16* base, int ld) {
  const int l = lane_id();
  const int m = l & 15;
  const int kb = (l >> 4) << 4;
  const _Float16* p = base + m * ld + kb;
  V16U u;
  u.h[0] = *(const v8h*)p;
  u.h[1] = *(const v8h*)(p + 8);
  return u.v;
}

// B operand from a pre-swizzled buffer: tile t occupies 512 halfs; lane's 16
// halfs are contiguous -> one aligned 32B load (2x ds_load_b128).
static __device__ __forceinline__ v16h lds_load_B16_swz(const _Float16* swz,
                                                        int tile) {
  return *(const v16h*)(swz + (tile << 9) + (lane_id() << 4));
}

// B operand that is the TRANSPOSE of a row-major f16 source:
// element (k, n) = src[(nBase + n) * ld + kOff + k]. Contiguous per lane.
static __device__ __forceinline__ v16h lds_load_B16_tr(const _Float16* src, int ld,
                                                       int nBase, int kOff) {
  const int l = lane_id();
  const int n = l & 15;
  const int kb = (l >> 4) << 4;
  return *(const v16h*)&src[(nBase + n) * ld + kOff + kb];
}

// B operand: 32x16 (KxN) f16 tile from row-major LDS (strided; used only for
// the 4 K-tiles in the rank-32 M update).
static __device__ __forceinline__ v16h lds_load_B16(const _Float16* base, int ld) {
  const int l = lane_id();
  const int n = l & 15;
  const int kb = (l >> 4) << 4;
  v16h b;
#pragma unroll
  for (int h = 0; h < 16; ++h) b[h] = base[(kb + h) * ld + n];
  return b;
}

// C/D layout helper: lane -> (mBase, n). VGPR j holds row mBase+j, column n.
static __device__ __forceinline__ void d_coords(int& mBase, int& n) {
  const int l = lane_id();
  n = l & 15;
  mBase = (l >> 4) << 3;
}

// Build a B-swizzled f16 copy of a row-major f32 matrix W (K x N).
// Tile index = kk * (N/16) + nt; within tile: lane l, half h ->
//   element (k = kk*32 + 16*(l>=16) + h, n = nt*16 + (l&15)).
static __device__ __forceinline__ void build_B_swz(const float* __restrict__ W,
                                                   _Float16* __restrict__ dst,
                                                   int K, int N, int tid,
                                                   int nthreads) {
  const int ntiles = N >> 4;
  const int total = K * N;
  for (int d = tid; d < total; d += nthreads) {
    const int tile = d >> 9;
    const int l = (d >> 4) & 31;
    const int h = d & 15;
    const int kk = tile / ntiles;
    const int nt = tile - kk * ntiles;
    const int k = (kk << 5) + ((l >> 4) << 4) + h;
    const int n = (nt << 4) + (l & 15);
    dst[d] = (_Float16)W[k * N + n];
  }
}

// Workspace layout (halfs): [hn: kB*kL*kH] [W1s: kH*kFF] [W2s: kFF*kH]
//                           [WkSw: kH*kH] [WvSw: kH*kH]
constexpr size_t kWsHn  = (size_t)kB * kL * kH;
constexpr size_t kWsW1  = kWsHn;
constexpr size_t kWsW2  = kWsW1 + (size_t)kH * kFF;
constexpr size_t kWsWk  = kWsW2 + (size_t)kFF * kH;
constexpr size_t kWsWv  = kWsWk + (size_t)kH * kH;

// ---------------------------------------------------------------------------
// Kernel P: build all B-swizzled weight images once (1 block, 256 threads).
// ---------------------------------------------------------------------------
__global__ __launch_bounds__(256) void prep_weights_kernel(
    const float* __restrict__ W1, const float* __restrict__ W2,
    const float* __restrict__ Wk, const float* __restrict__ Wv,
    _Float16* __restrict__ ws) {
  const int tid = (int)threadIdx.x;
  build_B_swz(W1, ws + kWsW1, kH, kFF, tid, 256);
  build_B_swz(W2, ws + kWsW2, kFF, kH, tid, 256);
  build_B_swz(Wk, ws + kWsWk, kH, kH, tid, 256);
  build_B_swz(Wv, ws + kWsWv, kH, kH, tid, 256);
}

// ---------------------------------------------------------------------------
// Kernel A: per-token  hn = LN(h + relu(h@W1+b1)@W2 + b2) * gamma + beta
// 4 waves/block, 16 tokens/wave, 64 tokens/block.
// ---------------------------------------------------------------------------
constexpr int kWavesA = 4;
constexpr int kTokPerBlock = kWavesA * 16;

__global__ __launch_bounds__(kWavesA * 32) void token_mlp_ln_kernel(
    const int* __restrict__ seq, const float* __restrict__ embed,
    const _Float16* __restrict__ wsWeights,  // swizzled W1s/W2s images
    const float* __restrict__ b1, const float* __restrict__ b2,
    const float* __restrict__ gamma, const float* __restrict__ beta,
    _Float16* __restrict__ hn_out) {
  __shared__ __align__(16) _Float16 sW1s[kH * kFF];  // 16 KB, B-swizzled
  __shared__ __align__(16) _Float16 sW2s[kFF * kH];  // 16 KB, B-swizzled
  __shared__ float sB1[kFF];
  __shared__ float sB2[kH];
  __shared__ float sG[kH];
  __shared__ float sBt[kH];
  __shared__ __align__(16) _Float16 sHt[kWavesA][16 * kH];   // h f16 -> hn f16
  __shared__ __align__(16) _Float16 sRt[kWavesA][16 * kFF];  // relu f16 -> x f32

  const int tid = (int)threadIdx.x;
  // Copy pre-swizzled weights: coalesced b128 loads -> b128 LDS stores.
  {
    const uint4* src1 = (const uint4*)(wsWeights + kWsW1);
    const uint4* src2 = (const uint4*)(wsWeights + kWsW2);
    uint4* dst1 = (uint4*)sW1s;
    uint4* dst2 = (uint4*)sW2s;
    for (int i = tid; i < (kH * kFF) / 8; i += kWavesA * 32) {
      dst1[i] = src1[i];
      dst2[i] = src2[i];
    }
  }
  for (int i = tid; i < kFF; i += kWavesA * 32) sB1[i] = b1[i];
  for (int i = tid; i < kH; i += kWavesA * 32) {
    sB2[i] = b2[i]; sG[i] = gamma[i]; sBt[i] = beta[i];
  }
  __syncthreads();

  const int wave = tid >> 5;
  const int lane = tid & 31;
  const long tok0 = (long)blockIdx.x * kTokPerBlock + wave * 16;

  // Gather 16 embedding rows (two lanes per row, 32 f32 each) -> f16 tile.
  {
    const int r = lane >> 1;
    const int c0 = (lane & 1) * 32;
    const int id = seq[tok0 + r];
    const float* erow = embed + (long)id * kH + c0;
#pragma unroll
    for (int c = 0; c < 32; ++c) sHt[wave][r * kH + c0 + c] = (_Float16)erow[c];
  }

  int mBase, nIdx;
  d_coords(mBase, nIdx);

  // Stage 1: R = relu(h @ W1 + b1)   (16x64 @ 64x128)
#pragma unroll
  for (int nt = 0; nt < kFF / 16; ++nt) {
    v8f acc = {};
#pragma unroll
    for (int kk = 0; kk < kH / 32; ++kk)
      acc = wmma32(lds_load_A16(&sHt[wave][kk * 32], kH),
                   lds_load_B16_swz(sW1s, kk * (kFF / 16) + nt), acc);
    const int n = nt * 16 + nIdx;
    const float bias = sB1[n];
#pragma unroll
    for (int j = 0; j < 8; ++j)
      sRt[wave][(mBase + j) * kFF + n] = (_Float16)fmaxf(acc[j] + bias, 0.f);
  }

  // Stage 2: x = h + R @ W2 + b2   (16x128 @ 128x64); keep all 4 tiles in regs
  // so sRt can be reused as the f32 x tile.
  v8f accs[kH / 16];
#pragma unroll
  for (int nt = 0; nt < kH / 16; ++nt) {
    v8f acc = {};
#pragma unroll
    for (int kk = 0; kk < kFF / 32; ++kk)
      acc = wmma32(lds_load_A16(&sRt[wave][kk * 32], kFF),
                   lds_load_B16_swz(sW2s, kk * (kH / 16) + nt), acc);
    accs[nt] = acc;
  }
  float* sXw = (float*)&sRt[wave][0];  // reuse: 16x128 f16 == 16x64 f32
#pragma unroll
  for (int nt = 0; nt < kH / 16; ++nt) {
    const int n = nt * 16 + nIdx;
    const float bias = sB2[n];
#pragma unroll
    for (int j = 0; j < 8; ++j) {
      const int row = mBase + j;
      sXw[row * kH + n] = (float)sHt[wave][row * kH + n] + accs[nt][j] + bias;
    }
  }

  // LayerNorm per row; write hn (f16) to LDS and to global workspace.
  {
    const int r = lane >> 1;
    const int c0 = (lane & 1) * 32;
    float s = 0.f, ss = 0.f;
#pragma unroll
    for (int c = 0; c < 32; ++c) {
      const float x = sXw[r * kH + c0 + c];
      s += x; ss += x * x;
    }
    s  += __shfl_xor(s, 1, 32);
    ss += __shfl_xor(ss, 1, 32);
    const float mu = s * (1.f / kH);
    const float var = ss * (1.f / kH) - mu * mu;
    const float rstd = rsqrtf(var + 1e-5f);
    float hnv[32];
#pragma unroll
    for (int c = 0; c < 32; ++c) {
      const int col = c0 + c;
      hnv[c] = (sXw[r * kH + col] - mu) * rstd * sG[col] + sBt[col];
      sHt[wave][r * kH + col] = (_Float16)hnv[c];
    }
    uint32_t* dst = (uint32_t*)(hn_out + (tok0 + r) * kH + c0);
#pragma unroll
    for (int c = 0; c < 16; ++c) {
      union { _Float16 h[2]; uint32_t u; } pk;
      pk.h[0] = (_Float16)hnv[2 * c];
      pk.h[1] = (_Float16)hnv[2 * c + 1];
      dst[c] = pk.u;
    }
  }
}

// ---------------------------------------------------------------------------
// Kernel B: chunked delta-rule scan, one wave32 per batch element.
//   per chunk (C=32):  K = hn@Wk (rows L2-normalized), V = hn@Wv,
//     Vp = K @ M0^T, A = K K^T, forward substitution -> U, M += U^T K.
// Then fused head: q = hn[L-1]@Wq, out = ((M q)@Wr + br)@Wo + bo.
// ---------------------------------------------------------------------------

// Issue one chunk's rows as async global->LDS b128 DMA (8 x 16B per lane).
// inst_offset advances both the global and LDS address in lockstep.
static __device__ __forceinline__ void issue_chunk_async(
    const _Float16* __restrict__ hnB, int t0, _Float16* buf) {
  const int lane = lane_id();
  const int t = t0 + lane;
  if (t < kL - 1) {
    const uint32_t ldsOff = (uint32_t)(uintptr_t)&buf[lane * kH];
    const _Float16* g = hnB + (size_t)t * kH;
#pragma unroll
    for (int j = 0; j < 8; ++j) {
      asm volatile("global_load_async_to_lds_b128 %0, %1, off offset:%2"
                   :: "v"(ldsOff), "v"(g), "i"(j * 16)
                   : "memory");
    }
  }
}

__global__ __launch_bounds__(32) void delta_scan_kernel(
    const _Float16* __restrict__ hn, const _Float16* __restrict__ wsWeights,
    const float* __restrict__ Wq, const float* __restrict__ Wr,
    const float* __restrict__ br, const float* __restrict__ Wo,
    const float* __restrict__ bo, float* __restrict__ out) {
  __shared__ __align__(16) float    sM[kH * kH];           // 16 KB  M[i][j]
  __shared__ __align__(16) _Float16 sMtSw[kH * kH];        //  8 KB  M^T, B-swizzled
  __shared__ __align__(16) _Float16 sWkSw[kH * kH];        //  8 KB  B-swizzled
  __shared__ __align__(16) _Float16 sWvSw[kH * kH];        //  8 KB  B-swizzled
  __shared__ __align__(16) _Float16 sHn[2][kChunk * kH];   //  8 KB  ping-pong
  __shared__ __align__(16) _Float16 sK[kChunk * kH];       //  4 KB  kn chunk
  __shared__ __align__(16) float    sU[kChunk * kH];       //  8 KB  v -> v - vp
  __shared__ __align__(16) float    sA[kChunk * kChunk];   //  4 KB  K K^T
  __shared__ float sVec[2 * kH];

  const int lane = (int)threadIdx.x;
  const int b = (int)blockIdx.x;
  const _Float16* hnB = hn + (size_t)b * kL * kH;

  // Kick off chunk 0 DMA before doing any setup work.
  issue_chunk_async(hnB, 0, sHn[0]);

  for (int i = lane; i < kH * kH; i += 32) sM[i] = 0.f;
  {
    const uint4* srck = (const uint4*)(wsWeights + kWsWk);
    const uint4* srcv = (const uint4*)(wsWeights + kWsWv);
    uint4* dstk = (uint4*)sWkSw;
    uint4* dstv = (uint4*)sWvSw;
    for (int i = lane; i < (kH * kH) / 8; i += 32) {
      dstk[i] = srck[i];
      dstv[i] = srcv[i];
    }
  }

  int mBase, nIdx;
  d_coords(mBase, nIdx);

  const int nChunks = (kL - 1 + kChunk - 1) / kChunk;  // 128
#pragma unroll 1
  for (int c = 0; c < nChunks; ++c) {
    const int t0 = c * kChunk;
    _Float16* cur = sHn[c & 1];

    // 1. Prefetch chunk c+1 into the other buffer, then wait for chunk c.
    //    Async loads complete in order, so <=8 outstanding means chunk c's
    //    8 transfers are done.
    if (c + 1 < nChunks) {
      issue_chunk_async(hnB, t0 + kChunk, sHn[(c + 1) & 1]);
      asm volatile("s_wait_asynccnt 0x8" ::: "memory");
    } else {
      asm volatile("s_wait_asynccnt 0x0" ::: "memory");
    }
    // Zero-pad rows t >= L-1 (k=0,v=0 rows are exact no-ops in the delta
    // rule); only the final chunk has one such row.
    if (t0 + lane >= kL - 1) {
      uint64_t* d64 = (uint64_t*)&cur[lane * kH];
#pragma unroll
      for (int j = 0; j < 16; ++j) d64[j] = 0ull;
    }

    // 2. K = hn @ Wk (f16), then L2-normalize each row.
#pragma unroll
    for (int mt = 0; mt < kChunk / 16; ++mt)
#pragma unroll
      for (int nt = 0; nt < kH / 16; ++nt) {
        v8f acc = {};
#pragma unroll
        for (int kk = 0; kk < kH / 32; ++kk)
          acc = wmma32(lds_load_A16(&cur[mt * 16 * kH + kk * 32], kH),
                       lds_load_B16_swz(sWkSw, kk * (kH / 16) + nt), acc);
        const int n = nt * 16 + nIdx;
#pragma unroll
        for (int j = 0; j < 8; ++j)
          sK[(mt * 16 + mBase + j) * kH + n] = (_Float16)acc[j];
      }
    {
      float ss = 0.f;
#pragma unroll
      for (int j = 0; j < kH; ++j) {
        const float kv = (float)sK[lane * kH + j];
        ss += kv * kv;
      }
      const float scale = 1.f / fmaxf(sqrtf(ss), 1e-12f);
#pragma unroll
      for (int j = 0; j < kH; ++j)
        sK[lane * kH + j] = (_Float16)((float)sK[lane * kH + j] * scale);
    }

    // 3. V = hn @ Wv  (f32 into sU).
#pragma unroll
    for (int mt = 0; mt < kChunk / 16; ++mt)
#pragma unroll
      for (int nt = 0; nt < kH / 16; ++nt) {
        v8f acc = {};
#pragma unroll
        for (int kk = 0; kk < kH / 32; ++kk)
          acc = wmma32(lds_load_A16(&cur[mt * 16 * kH + kk * 32], kH),
                       lds_load_B16_swz(sWvSw, kk * (kH / 16) + nt), acc);
        const int n = nt * 16 + nIdx;
#pragma unroll
        for (int j = 0; j < 8; ++j)
          sU[(mt * 16 + mBase + j) * kH + n] = acc[j];
      }

    // 4. Refresh M^T (f16) directly into the B-swizzled layout:
    //    tile (kk, nt): contraction index j = kk*32 + 16*(l>=16) + h,
    //                   output index   i = nt*16 + (l&15);  value = M[i][j].
    for (int d = lane; d < kH * kH; d += 32) {
      const int tile = d >> 9;
      const int l = (d >> 4) & 31;
      const int h = d & 15;
      const int kk = tile >> 2;
      const int nt = tile & 3;
      const int j = (kk << 5) + ((l >> 4) << 4) + h;
      const int i = (nt << 4) + (l & 15);
      sMtSw[d] = (_Float16)sM[i * kH + j];
    }

    // 5. sU -= K @ M0^T   (vp = M0 k).
#pragma unroll
    for (int mt = 0; mt < kChunk / 16; ++mt)
#pragma unroll
      for (int nt = 0; nt < kH / 16; ++nt) {
        v8f acc = {};
#pragma unroll
        for (int kk = 0; kk < kH / 32; ++kk)
          acc = wmma32(lds_load_A16(&sK[mt * 16 * kH + kk * 32], kH),
                       lds_load_B16_swz(sMtSw, kk * (kH / 16) + nt), acc);
        const int n = nt * 16 + nIdx;
#pragma unroll
        for (int j = 0; j < 8; ++j)
          sU[(mt * 16 + mBase + j) * kH + n] -= acc[j];
      }

    // 6. A = K K^T (32x32); B operand is K transposed (contiguous per lane).
#pragma unroll
    for (int mt = 0; mt < kChunk / 16; ++mt)
#pragma unroll
      for (int nt = 0; nt < kChunk / 16; ++nt) {
        v8f acc = {};
#pragma unroll
        for (int kk = 0; kk < kH / 32; ++kk)
          acc = wmma32(lds_load_A16(&sK[mt * 16 * kH + kk * 32], kH),
                       lds_load_B16_tr(sK, kH, nt * 16, kk * 32), acc);
        const int n = nt * 16 + nIdx;
#pragma unroll
        for (int j = 0; j < 8; ++j)
          sA[(mt * 16 + mBase + j) * kChunk + n] = acc[j];
      }

    // 7. Forward substitution entirely in registers; each lane owns columns
    //    (lane, lane+32):  u_t = (v_t - vp_t) - sum_{s<t} A[t][s] u_s.
    {
      float u0[kChunk], u1[kChunk];
#pragma unroll
      for (int t = 0; t < kChunk; ++t) {
        u0[t] = sU[t * kH + lane];
        u1[t] = sU[t * kH + lane + 32];
      }
#pragma unroll
      for (int t = 1; t < kChunk; ++t) {
#pragma unroll
        for (int s = 0; s < t; ++s) {
          const float w = sA[t * kChunk + s];
          u0[t] -= w * u0[s];
          u1[t] -= w * u1[s];
        }
      }
      // Write U^T (f16) into the *current* (now dead) hn buffer: row i of U^T
      // is contiguous in t -> vectorizable stores, and step 8 reads it as a
      // standard row-major A operand. The in-flight DMA targets the other
      // buffer, so there is no conflict.
      _Float16* sUT = cur;
#pragma unroll
      for (int t = 0; t < kChunk; ++t) {
        sUT[lane * kChunk + t] = (_Float16)u0[t];
        sUT[(lane + 32) * kChunk + t] = (_Float16)u1[t];
      }
    }

    // 8. M += U^T @ K   (64x32 @ 32x64, single K step; A = U^T row-major).
    {
      const _Float16* sUT = cur;
#pragma unroll
      for (int mt = 0; mt < kH / 16; ++mt)
#pragma unroll
        for (int nt = 0; nt < kH / 16; ++nt) {
          v8f acc = {};
          acc = wmma32(lds_load_A16(&sUT[mt * 16 * kChunk], kChunk),
                       lds_load_B16(&sK[nt * 16], kH), acc);
          const int n = nt * 16 + nIdx;
#pragma unroll
          for (int j = 0; j < 8; ++j)
            sM[(mt * 16 + mBase + j) * kH + n] += acc[j];
        }
    }
  }

  // ---- Fused output head (tiny; plain VALU, 2 elements per lane/stage) ----
  {
    const _Float16* last = hnB + (size_t)(kL - 1) * kH;
    for (int i = lane; i < kH; i += 32) sVec[i] = (float)last[i];
    // q = hn_last @ Wq
    float q0 = 0.f, q1 = 0.f;
    for (int i = 0; i < kH; ++i) {
      const float hv = sVec[i];
      q0 += hv * Wq[i * kH + lane];
      q1 += hv * Wq[i * kH + lane + 32];
    }
    sVec[kH + lane] = q0;
    sVec[kH + lane + 32] = q1;
    // y = M q
    float y0 = 0.f, y1 = 0.f;
    for (int j = 0; j < kH; ++j) {
      const float qv = sVec[kH + j];
      y0 += sM[lane * kH + j] * qv;
      y1 += sM[(lane + 32) * kH + j] * qv;
    }
    sVec[lane] = y0;
    sVec[lane + 32] = y1;
    // r = y @ Wr + br
    float r0 = br[lane], r1 = br[lane + 32];
    for (int i = 0; i < kH; ++i) {
      const float yv = sVec[i];
      r0 += yv * Wr[i * kH + lane];
      r1 += yv * Wr[i * kH + lane + 32];
    }
    sVec[kH + lane] = r0;
    sVec[kH + lane + 32] = r1;
    // out = r @ Wo + bo
    float o0 = bo[lane], o1 = bo[lane + 32];
    for (int j = 0; j < kH; ++j) {
      const float rv = sVec[kH + j];
      o0 += rv * Wo[j * kV + lane];
      o1 += rv * Wo[j * kV + lane + 32];
    }
    out[(size_t)b * kV + lane] = o0;
    out[(size_t)b * kV + lane + 32] = o1;
  }
}

// ---------------------------------------------------------------------------
extern "C" void kernel_launch(void* const* d_in, const int* in_sizes, int n_in,
                              void* d_out, int out_size, void* d_ws, size_t ws_size,
                              hipStream_t stream) {
  (void)in_sizes; (void)n_in; (void)out_size; (void)ws_size;
  const int*   seq   = (const int*)  d_in[0];
  const float* embed = (const float*)d_in[1];
  const float* W1    = (const float*)d_in[2];
  const float* b1    = (const float*)d_in[3];
  const float* W2    = (const float*)d_in[4];
  const float* b2    = (const float*)d_in[5];
  const float* gamma = (const float*)d_in[6];
  const float* beta  = (const float*)d_in[7];
  const float* Wk    = (const float*)d_in[8];
  const float* Wv    = (const float*)d_in[9];
  const float* Wq    = (const float*)d_in[10];
  const float* Wr    = (const float*)d_in[11];
  const float* br    = (const float*)d_in[12];
  const float* Wo    = (const float*)d_in[13];
  const float* bo    = (const float*)d_in[14];
  float* out = (float*)d_out;

  _Float16* ws = (_Float16*)d_ws;  // [hn | W1s | W2s | WkSw | WvSw]

  prep_weights_kernel<<<dim3(1), dim3(256), 0, stream>>>(W1, W2, Wk, Wv, ws);

  const long nTok = (long)kB * kL;
  const unsigned gridA = (unsigned)(nTok / kTokPerBlock);
  token_mlp_ln_kernel<<<dim3(gridA), dim3(kWavesA * 32), 0, stream>>>(
      seq, embed, ws, b1, b2, gamma, beta, ws /* hn at offset 0 */);
  delta_scan_kernel<<<dim3(kB), dim3(32), 0, stream>>>(
      ws /* hn */, ws /* weights */, Wq, Wr, br, Wo, bo, out);
}
